// NMPEncoder_34385508171832
// MI455X (gfx1250) — matrix operand order repaired and checked by reading, not compiled, not run on hardware
//
#include <hip/hip_runtime.h>
#include <math.h>

// ---------------------------------------------------------------------------
// MI455X (gfx1250) NMP encoder.
//   * all GEMMs run on V_WMMA_F32_16X16X32_BF16 (wave32, f32 accumulate)
//   * GEMM staging uses GLOBAL_LOAD_ASYNC_TO_LDS_B128 (ASYNCcnt) -> no VGPR
//     round-trip, no loadcnt stall before LDS store; OOB slots pre-zeroed once
//   * node2edge gathers are fused into the GEMM A-operand loader
//   * every GEMM K is a multiple of 32 (sem K=300 zero-padded to 320)
//   * BatchNorm = column stats kernel + in-place apply (bf16)
//   * two 66.6 MB bf16 edge buffers ping-pong; total ws ~160 MB
// Input pointer order assumes jax sorted-dict flattening of setup_inputs():
//   0 inputs, 1 spatial_feats, 2 rel_rec, 3 rel_send, 4 bbox_loc,
//   5 cls_W, 6 cls_b, 7..12 e2n{W1,W2,b1,b2,beta,g}, 13 fus_W, 14 fus_b,
//   15..20 mlp1, 21..26 mlp2, 27..32 mlp3, 33..38 od, 39 rel_W, 40 rel_b,
//   41..46 sem, 47..52 vis.
// ---------------------------------------------------------------------------

typedef __bf16 bf16_t;
typedef __attribute__((ext_vector_type(16))) __bf16 v16bf;
typedef __attribute__((ext_vector_type(8)))  float  v8f;
typedef __attribute__((ext_vector_type(4)))  unsigned u32x4;

#define B_    8
#define N_    128
#define NM1   127
#define E_    16256          // N*(N-1)
#define M_NODE 1024          // B*N
#define M_EDGE 130048        // B*E

#define TILE_M 64
#define TILE_N 64
#define TILE_K 32
#define ASTRIDE 56           // bf16 units; 112-byte rows (16B aligned, bank-spread)

// async copy of 16 bytes global -> LDS, tracked by ASYNCcnt (cdna5 08_async §4)
__device__ __forceinline__ void async_copy16(void* lds_dst, const void* gsrc) {
  const unsigned ldsAddr = (unsigned)(size_t)lds_dst;          // low 32 bits = LDS offset
  const unsigned long long ga = (unsigned long long)(size_t)gsrc;
  asm volatile("global_load_async_to_lds_b128 %0, %1, off"
               :: "v"(ldsAddr), "v"(ga) : "memory");
}
__device__ __forceinline__ void wait_async0() {
  asm volatile("s_wait_asynccnt 0x0" ::: "memory");
}

// ------------------------------ GEMM ---------------------------------------
// C[M,Nc] = act(A[M,K] @ Wt[Nc,K]^T + bias)   (Wt is pre-transposed bf16)
// Requires: K % 32 == 0 (caller pads), all row bases 16B aligned.
// MODE 0: A is a plain bf16 row-major matrix.
// MODE 1: A row m = concat(nodeF[b,i,:256], nodeF[b,j,:256])  (node2edge)
// MODE 2: A row m = concat(h[b,i], h[b,j], edgeF[m,:256])     (node2edge + skip)
// ACT: 0 none, 1 ELU, 2 ReLU.  OUTBF: 1 -> bf16 C, 0 -> f32 C.
template <int MODE, int ACT, int OUTBF>
__global__ void __launch_bounds__(256)
gemm_wmma(const bf16_t* __restrict__ A, const bf16_t* __restrict__ Wt,
          const float* __restrict__ bias, void* __restrict__ Cptr,
          int M, int Nc, int K,
          const bf16_t* __restrict__ S0, const bf16_t* __restrict__ S1) {
  __shared__ __align__(16) bf16_t As[TILE_M * ASTRIDE];
  __shared__ __align__(16) bf16_t Bs[TILE_N * ASTRIDE];

  const int tid  = threadIdx.x;
  const int lane = tid & 31;
  const int wid  = tid >> 5;
  const int wm   = wid & 3;        // 4 wave-rows of 16
  const int wn   = wid >> 2;       // 2 wave-cols of 32
  const int m0   = blockIdx.y * TILE_M;
  const int n0   = blockIdx.x * TILE_N;
  const int h    = lane >> 4;      // lane half (0/1)
  const int r    = lane & 15;

  // ---- per-thread staging assignment: row = tid/4, 8 elems at k = (tid%4)*8
  const int lrow = tid >> 2;
  const int lk   = (tid & 3) * 8;
  bf16_t* aSlot = &As[lrow * ASTRIDE + lk];
  bf16_t* bSlot = &Bs[lrow * ASTRIDE + lk];

  // ---- hoisted gather bases for the A row this thread stages
  const int am = m0 + lrow;
  const bool amOK = (am < M);
  const bf16_t* aRow0 = nullptr;
  const bf16_t* aRecv = nullptr;
  const bf16_t* aSend = nullptr;
  const bf16_t* aEdge = nullptr;
  if (amOK) {
    if (MODE == 0) {
      aRow0 = A + (size_t)am * K;
    } else {
      const int b = am / E_;
      const int e = am - b * E_;
      const int i = e / NM1;
      const int rr = e - i * NM1;
      const int j = rr + (rr >= i ? 1 : 0);       // skip self-loop
      aRecv = S0 + ((size_t)(b * N_ + i)) * 256;
      aSend = S0 + ((size_t)(b * N_ + j)) * 256;
      if (MODE == 2) aEdge = S1 + (size_t)am * 256;
    }
  }
  // segment-select for the 8-element group starting at k (group never straddles)
  auto aPtr = [&](int k) -> const bf16_t* {
    if (MODE == 0) return aRow0 + k;
    if (k < 256)   return aRecv + k;
    if (MODE == 1) return aSend + (k - 256);
    if (k < 512)   return aSend + (k - 256);
    return aEdge + (k - 512);
  };
  const int     bnr  = n0 + lrow;
  const bool    bnOK = (bnr < Nc);
  const bf16_t* bRow = bnOK ? (Wt + (size_t)bnr * K) : nullptr;

  // ---- OOB staging slots are loop-invariant: zero them once
  {
    const u32x4 z = {0u, 0u, 0u, 0u};
    if (!amOK) *(u32x4*)aSlot = z;
    if (!bnOK) *(u32x4*)bSlot = z;
  }

  v8f acc0 = {};
  v8f acc1 = {};

  for (int kb = 0; kb < K; kb += TILE_K) {
    // ---- async-stage A/B tiles straight into LDS (no VGPR round-trip)
    if (amOK) async_copy16(aSlot, aPtr(kb + lk));
    if (bnOK) async_copy16(bSlot, bRow + kb + lk);
    // ---- prefetch next K-tile global lines (global_prefetch_b8)
    if (kb + TILE_K < K) {
      if (amOK) __builtin_prefetch((const void*)aPtr(kb + TILE_K + lk), 0, 1);
      if (bnOK) __builtin_prefetch((const void*)(bRow + kb + TILE_K + lk), 0, 1);
    }
    wait_async0();           // our lanes' copies landed in LDS
    __syncthreads();         // all waves' copies landed

    // ---- build fragments per ISA 7.12.2 16-bit layouts
    union { v16bf v; unsigned u[8]; } af, bf0, bf1;
    const unsigned* Arow = (const unsigned*)&As[(wm * 16 + r) * ASTRIDE];
#pragma unroll
    for (int d = 0; d < 8; ++d) {
      const int k = (d < 4 ? 2 * d : 16 + 2 * (d - 4)) + h * 8;  // A: K pairs
      af.u[d] = Arow[k >> 1];
    }
    const unsigned* Brow0 = (const unsigned*)&Bs[(wn * 32 + r) * ASTRIDE];
    const unsigned* Brow1 = (const unsigned*)&Bs[(wn * 32 + 16 + r) * ASTRIDE];
#pragma unroll
    for (int d = 0; d < 8; ++d) {
      const int k = (d < 4 ? 2 * d : 16 + 2 * (d - 4)) + h * 8;  // B: K pairs
      bf0.u[d] = Brow0[k >> 1];
      bf1.u[d] = Brow1[k >> 1];
    }

    acc0 = __builtin_amdgcn_wmma_f32_16x16x32_bf16(false, af.v, false, bf0.v,
                                                   (short)0, acc0, false, false);
    acc1 = __builtin_amdgcn_wmma_f32_16x16x32_bf16(false, af.v, false, bf1.v,
                                                   (short)0, acc1, false, false);
    __syncthreads();         // fragments consumed; LDS reusable next iter
  }

  // ---- epilogue: bias + activation + store (C layout: VGPR q -> M = h*8+q, N = r)
  const int nA = n0 + wn * 32 + r;
  const int nB = nA + 16;
  const float biasA = (nA < Nc) ? bias[nA] : 0.0f;
  const float biasB = (nB < Nc) ? bias[nB] : 0.0f;
#pragma unroll
  for (int q = 0; q < 8; ++q) {
    const int mr = m0 + wm * 16 + h * 8 + q;
    if (mr >= M) continue;
    float v0 = acc0[q] + biasA;
    float v1 = acc1[q] + biasB;
    if (ACT == 1) {
      v0 = v0 > 0.0f ? v0 : expm1f(v0);
      v1 = v1 > 0.0f ? v1 : expm1f(v1);
    } else if (ACT == 2) {
      v0 = fmaxf(v0, 0.0f);
      v1 = fmaxf(v1, 0.0f);
    }
    if (OUTBF) {
      bf16_t* C = (bf16_t*)Cptr;
      if (nA < Nc) C[(size_t)mr * Nc + nA] = (bf16_t)v0;
      if (nB < Nc) C[(size_t)mr * Nc + nB] = (bf16_t)v1;
    } else {
      float* C = (float*)Cptr;
      if (nA < Nc) C[(size_t)mr * Nc + nA] = v0;
      if (nB < Nc) C[(size_t)mr * Nc + nB] = v1;
    }
  }
}

// ------------------------- support kernels ---------------------------------

// f32 [rows, srcLd] slice -> bf16 [rows, dstW], zero-padding cols >= width
__global__ void cvt_slice(const float* __restrict__ src, bf16_t* __restrict__ dst,
                          int rows, int srcLd, int srcOff, int width, int dstW) {
  const size_t idx = (size_t)blockIdx.x * blockDim.x + threadIdx.x;
  if (idx >= (size_t)rows * dstW) return;
  const int c = (int)(idx % dstW);
  const size_t m = idx / dstW;
  dst[idx] = (c < width) ? (bf16_t)src[m * srcLd + srcOff + c] : (bf16_t)0.0f;
}

// f32 W[K,Nc] -> bf16 Wt[Nc,Kp], zero-padding k >= K
__global__ void wt_transpose(const float* __restrict__ W, bf16_t* __restrict__ Wt,
                             int K, int Kp, int Nc) {
  const size_t idx = (size_t)blockIdx.x * blockDim.x + threadIdx.x;
  if (idx >= (size_t)Nc * Kp) return;
  const int k = (int)(idx % Kp);
  const size_t n = idx / Kp;
  Wt[idx] = (k < K) ? (bf16_t)W[(size_t)k * Nc + n] : (bf16_t)0.0f;
}

// column mean & rsqrt(var+eps) over M rows of bf16 [M,D]; one block per column
__global__ void bn_stats(const bf16_t* __restrict__ X, int M, int D,
                         float* __restrict__ mu, float* __restrict__ rsig) {
  const int d = blockIdx.x;
  float s = 0.0f, s2 = 0.0f;
  for (int m = threadIdx.x; m < M; m += blockDim.x) {
    const float v = (float)X[(size_t)m * D + d];
    s += v; s2 += v * v;
  }
  __shared__ float sh[256], sh2[256];
  sh[threadIdx.x] = s; sh2[threadIdx.x] = s2;
  __syncthreads();
  for (int st = 128; st > 0; st >>= 1) {
    if (threadIdx.x < st) {
      sh[threadIdx.x]  += sh[threadIdx.x + st];
      sh2[threadIdx.x] += sh2[threadIdx.x + st];
    }
    __syncthreads();
  }
  if (threadIdx.x == 0) {
    const float m_ = sh[0] / (float)M;
    const float var = sh2[0] / (float)M - m_ * m_;
    mu[d]   = m_;
    rsig[d] = rsqrtf(var + 1e-5f);
  }
}

// y = (x - mu)*rsig*g + beta   (bf16 in, bf16 out at [m*ldY + colOff + d])
__global__ void bn_apply(const bf16_t* __restrict__ X, bf16_t* __restrict__ Y,
                         int M, int D, const float* __restrict__ mu,
                         const float* __restrict__ rsig, const float* __restrict__ g,
                         const float* __restrict__ beta, int ldY, int colOff) {
  const size_t idx = (size_t)blockIdx.x * blockDim.x + threadIdx.x;
  if (idx >= (size_t)M * D) return;
  const int d = (int)(idx % D);
  const size_t m = idx / D;
  float v = (float)X[idx];
  v = (v - mu[d]) * rsig[d] * g[d] + beta[d];
  Y[m * (size_t)ldY + colOff + d] = (bf16_t)v;
}

// edge2node: incoming/outgoing degree-normalized sums over edge_feats bf16 [B*E,256]
// -> out bf16 [B*N, 512]  (cols 0..255 incoming mean, 256..511 outgoing mean)
__global__ void edge2node_k(const bf16_t* __restrict__ EF, bf16_t* __restrict__ out) {
  const int idx = blockIdx.x * blockDim.x + threadIdx.x;
  if (idx >= B_ * N_ * 256) return;
  const int c  = idx & 255;
  const int bn = idx >> 8;
  const int b  = bn >> 7;
  const int n  = bn & 127;
  const size_t base = (size_t)b * E_ * 256;
  // receiver-n edges are a contiguous block of 127 rows
  float inc = 0.0f;
  const size_t rb = base + (size_t)n * NM1 * 256 + c;
  for (int t = 0; t < NM1; ++t) inc += (float)EF[rb + (size_t)t * 256];
  // sender-n edges are strided
  float outg = 0.0f;
  for (int i = 0; i < N_; ++i) {
    if (i == n) continue;
    const int e = i * NM1 + (n < i ? n : n - 1);
    outg += (float)EF[base + (size_t)e * 256 + c];
  }
  const float s = 1.0f / (float)NM1;
  out[(size_t)bn * 512 + c]       = (bf16_t)(inc * s);
  out[(size_t)bn * 512 + 256 + c] = (bf16_t)(outg * s);
}

// ------------------------------ launch -------------------------------------

static inline dim3 gemm_grid(int M, int Nc) {
  return dim3((Nc + TILE_N - 1) / TILE_N, (M + TILE_M - 1) / TILE_M);
}

extern "C" void kernel_launch(void* const* d_in, const int* in_sizes, int n_in,
                              void* d_out, int out_size, void* d_ws, size_t ws_size,
                              hipStream_t stream) {
  (void)in_sizes; (void)n_in; (void)out_size; (void)ws_size;

  // ---- input pointers (sorted-dict flattening, see header comment)
  const float* x        = (const float*)d_in[0];
  const float* cls_W    = (const float*)d_in[5];
  const float* cls_b    = (const float*)d_in[6];
  const float* e2n_W1   = (const float*)d_in[7];
  const float* e2n_W2   = (const float*)d_in[8];
  const float* e2n_b1   = (const float*)d_in[9];
  const float* e2n_b2   = (const float*)d_in[10];
  const float* e2n_beta = (const float*)d_in[11];
  const float* e2n_g    = (const float*)d_in[12];
  const float* fus_W    = (const float*)d_in[13];
  const float* fus_b    = (const float*)d_in[14];
  const float* m1_W1    = (const float*)d_in[15];
  const float* m1_W2    = (const float*)d_in[16];
  const float* m1_b1    = (const float*)d_in[17];
  const float* m1_b2    = (const float*)d_in[18];
  const float* m1_beta  = (const float*)d_in[19];
  const float* m1_g     = (const float*)d_in[20];
  const float* m2_W1    = (const float*)d_in[21];
  const float* m2_W2    = (const float*)d_in[22];
  const float* m2_b1    = (const float*)d_in[23];
  const float* m2_b2    = (const float*)d_in[24];
  const float* m2_beta  = (const float*)d_in[25];
  const float* m2_g     = (const float*)d_in[26];
  const float* m3_W1    = (const float*)d_in[27];
  const float* m3_W2    = (const float*)d_in[28];
  const float* m3_b1    = (const float*)d_in[29];
  const float* m3_b2    = (const float*)d_in[30];
  const float* m3_beta  = (const float*)d_in[31];
  const float* m3_g     = (const float*)d_in[32];
  const float* od_W1    = (const float*)d_in[33];
  const float* od_W2    = (const float*)d_in[34];
  const float* od_b1    = (const float*)d_in[35];
  const float* od_b2    = (const float*)d_in[36];
  const float* od_beta  = (const float*)d_in[37];
  const float* od_g     = (const float*)d_in[38];
  const float* rel_W    = (const float*)d_in[39];
  const float* rel_b    = (const float*)d_in[40];
  const float* sem_W1   = (const float*)d_in[41];
  const float* sem_W2   = (const float*)d_in[42];
  const float* sem_b1   = (const float*)d_in[43];
  const float* sem_b2   = (const float*)d_in[44];
  const float* sem_beta = (const float*)d_in[45];
  const float* sem_g    = (const float*)d_in[46];
  const float* vis_W1   = (const float*)d_in[47];
  const float* vis_W2   = (const float*)d_in[48];
  const float* vis_b1   = (const float*)d_in[49];
  const float* vis_b2   = (const float*)d_in[50];
  const float* vis_beta = (const float*)d_in[51];
  const float* vis_g    = (const float*)d_in[52];

  float* out_rel = (float*)d_out;                       // [B,E,71]
  float* out_cls = out_rel + (size_t)B_ * E_ * 71;      // [B,N,101]

  // ---- workspace bump allocator (~160 MB)
  char* wsp = (char*)d_ws;
  auto alloc = [&](size_t bytes) -> void* {
    void* p = (void*)wsp;
    wsp += (bytes + 255) & ~(size_t)255;
    return p;
  };
  bf16_t* xv    = (bf16_t*)alloc((size_t)M_NODE * 4096 * 2);
  bf16_t* xd    = (bf16_t*)alloc((size_t)M_NODE * 4096 * 2);
  bf16_t* xs    = (bf16_t*)alloc((size_t)M_NODE * 320 * 2);   // K padded 300->320
  bf16_t* clsWt = (bf16_t*)alloc((size_t)101 * 4096 * 2);
  bf16_t* visW1t= (bf16_t*)alloc((size_t)128 * 4096 * 2);
  bf16_t* visW2t= (bf16_t*)alloc((size_t)128 * 128 * 2);
  bf16_t* odW1t = (bf16_t*)alloc((size_t)128 * 4096 * 2);
  bf16_t* odW2t = (bf16_t*)alloc((size_t)128 * 128 * 2);
  bf16_t* semW1t= (bf16_t*)alloc((size_t)256 * 320 * 2);      // K padded
  bf16_t* semW2t= (bf16_t*)alloc((size_t)256 * 256 * 2);
  bf16_t* fusWt = (bf16_t*)alloc((size_t)256 * 512 * 2);
  bf16_t* m1W1t = (bf16_t*)alloc((size_t)256 * 512 * 2);
  bf16_t* m1W2t = (bf16_t*)alloc((size_t)256 * 256 * 2);
  bf16_t* e2nW1t= (bf16_t*)alloc((size_t)256 * 512 * 2);
  bf16_t* e2nW2t= (bf16_t*)alloc((size_t)256 * 256 * 2);
  bf16_t* m2W1t = (bf16_t*)alloc((size_t)256 * 256 * 2);
  bf16_t* m2W2t = (bf16_t*)alloc((size_t)256 * 256 * 2);
  bf16_t* m3W1t = (bf16_t*)alloc((size_t)256 * 768 * 2);
  bf16_t* m3W2t = (bf16_t*)alloc((size_t)256 * 256 * 2);
  bf16_t* relWt = (bf16_t*)alloc((size_t)71 * 256 * 2);
  bf16_t* hid1  = (bf16_t*)alloc((size_t)M_NODE * 256 * 2);   // node hidden
  bf16_t* raw2  = (bf16_t*)alloc((size_t)M_NODE * 256 * 2);   // node L2 raw (pre-BN)
  bf16_t* ncat  = (bf16_t*)alloc((size_t)M_NODE * 512 * 2);   // concat(v,d,s) post-BN
  bf16_t* nodeF = (bf16_t*)alloc((size_t)M_NODE * 256 * 2);   // fused node feats
  bf16_t* e2nin = (bf16_t*)alloc((size_t)M_NODE * 512 * 2);   // edge2node output
  bf16_t* hbuf  = (bf16_t*)alloc((size_t)M_NODE * 256 * 2);   // h
  float*  mu    = (float*)alloc(256 * 4);
  float*  rsig  = (float*)alloc(256 * 4);
  bf16_t* bufA  = (bf16_t*)alloc((size_t)M_EDGE * 256 * 2);   // edge ping
  bf16_t* bufB  = (bf16_t*)alloc((size_t)M_EDGE * 256 * 2);   // edge pong

  const dim3 blk(256);
  auto g1 = [](size_t n) { return dim3((unsigned)((n + 255) / 256)); };

  // ---- 1. convert input slices to bf16 (sem slice zero-padded to 320 cols)
  cvt_slice<<<g1((size_t)M_NODE * 4096), blk, 0, stream>>>(x, xv, M_NODE, 8492, 0, 4096, 4096);
  cvt_slice<<<g1((size_t)M_NODE * 4096), blk, 0, stream>>>(x, xd, M_NODE, 8492, 4096, 4096, 4096);
  cvt_slice<<<g1((size_t)M_NODE * 320),  blk, 0, stream>>>(x, xs, M_NODE, 8492, 8192, 300, 320);

  // ---- 2. transpose-convert weights [K,N] f32 -> [N,Kp] bf16
  wt_transpose<<<g1((size_t)4096 * 101), blk, 0, stream>>>(cls_W, clsWt, 4096, 4096, 101);
  wt_transpose<<<g1((size_t)4096 * 128), blk, 0, stream>>>(vis_W1, visW1t, 4096, 4096, 128);
  wt_transpose<<<g1((size_t)128 * 128),  blk, 0, stream>>>(vis_W2, visW2t, 128, 128, 128);
  wt_transpose<<<g1((size_t)4096 * 128), blk, 0, stream>>>(od_W1, odW1t, 4096, 4096, 128);
  wt_transpose<<<g1((size_t)128 * 128),  blk, 0, stream>>>(od_W2, odW2t, 128, 128, 128);
  wt_transpose<<<g1((size_t)320 * 256),  blk, 0, stream>>>(sem_W1, semW1t, 300, 320, 256);
  wt_transpose<<<g1((size_t)256 * 256),  blk, 0, stream>>>(sem_W2, semW2t, 256, 256, 256);
  wt_transpose<<<g1((size_t)512 * 256),  blk, 0, stream>>>(fus_W, fusWt, 512, 512, 256);
  wt_transpose<<<g1((size_t)512 * 256),  blk, 0, stream>>>(m1_W1, m1W1t, 512, 512, 256);
  wt_transpose<<<g1((size_t)256 * 256),  blk, 0, stream>>>(m1_W2, m1W2t, 256, 256, 256);
  wt_transpose<<<g1((size_t)512 * 256),  blk, 0, stream>>>(e2n_W1, e2nW1t, 512, 512, 256);
  wt_transpose<<<g1((size_t)256 * 256),  blk, 0, stream>>>(e2n_W2, e2nW2t, 256, 256, 256);
  wt_transpose<<<g1((size_t)256 * 256),  blk, 0, stream>>>(m2_W1, m2W1t, 256, 256, 256);
  wt_transpose<<<g1((size_t)256 * 256),  blk, 0, stream>>>(m2_W2, m2W2t, 256, 256, 256);
  wt_transpose<<<g1((size_t)768 * 256),  blk, 0, stream>>>(m3_W1, m3W1t, 768, 768, 256);
  wt_transpose<<<g1((size_t)256 * 256),  blk, 0, stream>>>(m3_W2, m3W2t, 256, 256, 256);
  wt_transpose<<<g1((size_t)256 * 71),   blk, 0, stream>>>(rel_W, relWt, 256, 256, 71);

  // ---- 3. x_cls (f32 output straight into d_out tail)
  gemm_wmma<0,0,0><<<gemm_grid(M_NODE,101), blk, 0, stream>>>(
      xv, clsWt, cls_b, out_cls, M_NODE, 101, 4096, nullptr, nullptr);

  // ---- 4. three node MLPs -> ncat[1024,512]
  // vis
  gemm_wmma<0,1,1><<<gemm_grid(M_NODE,128), blk, 0, stream>>>(
      xv, visW1t, vis_b1, hid1, M_NODE, 128, 4096, nullptr, nullptr);
  gemm_wmma<0,1,1><<<gemm_grid(M_NODE,128), blk, 0, stream>>>(
      hid1, visW2t, vis_b2, raw2, M_NODE, 128, 128, nullptr, nullptr);
  bn_stats<<<dim3(128), blk, 0, stream>>>(raw2, M_NODE, 128, mu, rsig);
  bn_apply<<<g1((size_t)M_NODE*128), blk, 0, stream>>>(raw2, ncat, M_NODE, 128,
      mu, rsig, vis_g, vis_beta, 512, 0);
  // od
  gemm_wmma<0,1,1><<<gemm_grid(M_NODE,128), blk, 0, stream>>>(
      xd, odW1t, od_b1, hid1, M_NODE, 128, 4096, nullptr, nullptr);
  gemm_wmma<0,1,1><<<gemm_grid(M_NODE,128), blk, 0, stream>>>(
      hid1, odW2t, od_b2, raw2, M_NODE, 128, 128, nullptr, nullptr);
  bn_stats<<<dim3(128), blk, 0, stream>>>(raw2, M_NODE, 128, mu, rsig);
  bn_apply<<<g1((size_t)M_NODE*128), blk, 0, stream>>>(raw2, ncat, M_NODE, 128,
      mu, rsig, od_g, od_beta, 512, 128);
  // sem (K padded to 320)
  gemm_wmma<0,1,1><<<gemm_grid(M_NODE,256), blk, 0, stream>>>(
      xs, semW1t, sem_b1, hid1, M_NODE, 256, 320, nullptr, nullptr);
  gemm_wmma<0,1,1><<<gemm_grid(M_NODE,256), blk, 0, stream>>>(
      hid1, semW2t, sem_b2, raw2, M_NODE, 256, 256, nullptr, nullptr);
  bn_stats<<<dim3(256), blk, 0, stream>>>(raw2, M_NODE, 256, mu, rsig);
  bn_apply<<<g1((size_t)M_NODE*256), blk, 0, stream>>>(raw2, ncat, M_NODE, 256,
      mu, rsig, sem_g, sem_beta, 512, 256);

  // ---- 5. fusion: relu(ncat @ fus_W + fus_b) -> nodeF
  gemm_wmma<0,2,1><<<gemm_grid(M_NODE,256), blk, 0, stream>>>(
      ncat, fusWt, fus_b, nodeF, M_NODE, 256, 512, nullptr, nullptr);

  // ---- 6. mlp1 over edges (node2edge fused into A-loader) -> edge_feats=bufB
  gemm_wmma<1,1,1><<<gemm_grid(M_EDGE,256), blk, 0, stream>>>(
      nullptr, m1W1t, m1_b1, bufA, M_EDGE, 256, 512, nodeF, nullptr);
  gemm_wmma<0,1,1><<<gemm_grid(M_EDGE,256), blk, 0, stream>>>(
      bufA, m1W2t, m1_b2, bufB, M_EDGE, 256, 256, nullptr, nullptr);
  bn_stats<<<dim3(256), blk, 0, stream>>>(bufB, M_EDGE, 256, mu, rsig);
  bn_apply<<<g1((size_t)M_EDGE*256), blk, 0, stream>>>(bufB, bufB, M_EDGE, 256,
      mu, rsig, m1_g, m1_beta, 256, 0);

  // ---- 7. edge2node -> e2nin[1024,512]
  edge2node_k<<<g1((size_t)B_*N_*256), blk, 0, stream>>>(bufB, e2nin);

  // ---- 8. e2n MLP then mlp2 -> h (hbuf)
  gemm_wmma<0,1,1><<<gemm_grid(M_NODE,256), blk, 0, stream>>>(
      e2nin, e2nW1t, e2n_b1, hid1, M_NODE, 256, 512, nullptr, nullptr);
  gemm_wmma<0,1,1><<<gemm_grid(M_NODE,256), blk, 0, stream>>>(
      hid1, e2nW2t, e2n_b2, raw2, M_NODE, 256, 256, nullptr, nullptr);
  bn_stats<<<dim3(256), blk, 0, stream>>>(raw2, M_NODE, 256, mu, rsig);
  bn_apply<<<g1((size_t)M_NODE*256), blk, 0, stream>>>(raw2, hbuf, M_NODE, 256,
      mu, rsig, e2n_g, e2n_beta, 256, 0);
  gemm_wmma<0,1,1><<<gemm_grid(M_NODE,256), blk, 0, stream>>>(
      hbuf, m2W1t, m2_b1, hid1, M_NODE, 256, 256, nullptr, nullptr);
  gemm_wmma<0,1,1><<<gemm_grid(M_NODE,256), blk, 0, stream>>>(
      hid1, m2W2t, m2_b2, raw2, M_NODE, 256, 256, nullptr, nullptr);
  bn_stats<<<dim3(256), blk, 0, stream>>>(raw2, M_NODE, 256, mu, rsig);
  bn_apply<<<g1((size_t)M_NODE*256), blk, 0, stream>>>(raw2, hbuf, M_NODE, 256,
      mu, rsig, m2_g, m2_beta, 256, 0);

  // ---- 9. mlp3: concat(node2edge(h), edge_feats) fused in loader -> ef=bufB
  gemm_wmma<2,1,1><<<gemm_grid(M_EDGE,256), blk, 0, stream>>>(
      nullptr, m3W1t, m3_b1, bufA, M_EDGE, 256, 768, hbuf, bufB);
  gemm_wmma<0,1,1><<<gemm_grid(M_EDGE,256), blk, 0, stream>>>(
      bufA, m3W2t, m3_b2, bufB, M_EDGE, 256, 256, nullptr, nullptr);
  bn_stats<<<dim3(256), blk, 0, stream>>>(bufB, M_EDGE, 256, mu, rsig);
  bn_apply<<<g1((size_t)M_EDGE*256), blk, 0, stream>>>(bufB, bufB, M_EDGE, 256,
      mu, rsig, m3_g, m3_beta, 256, 0);

  // ---- 10. relation head -> d_out (f32)
  gemm_wmma<0,0,0><<<gemm_grid(M_EDGE,71), blk, 0, stream>>>(
      bufB, relWt, rel_b, out_rel, M_EDGE, 71, 256, nullptr, nullptr);
}
